// Attention_14353780703698
// MI455X (gfx1250) — matrix operand order repaired and checked by reading, not compiled, not run on hardware
//
#include <hip/hip_runtime.h>
#include <hip/hip_bf16.h>

typedef __attribute__((ext_vector_type(16))) _Float16 v16h;
typedef __attribute__((ext_vector_type(8)))  _Float16 v8h;
typedef __attribute__((ext_vector_type(8)))  float    v8f;
typedef __attribute__((ext_vector_type(4)))  float    v4f;

#define BB 2
#define SS 2048
#define CC 256
#define HH 8
#define DHD 32

// ---- raw (un-converted) fragment data ----------------------------------
struct Raw16f { v4f a, b, c, d; };   // 16 floats
struct Raw16h { v8h lo, hi; };       // 16 halfs

// A-matrix 16x32 f16 (ISA 7.12.2): lane<16 holds K=0..7 (e0..7), K=16..23 (e8..15);
// lane>=16 holds K=8..15, K=24..31.
__device__ __forceinline__ Raw16h raw_a_h16(const _Float16* base, int hi) {
  Raw16h r;
  r.lo = *(const v8h*)(base + 8 * hi);
  r.hi = *(const v8h*)(base + 16 + 8 * hi);
  return r;
}
// B-matrix 32x16 f16: lane<16 holds K=0..7, K=16..23; lane>=16 holds K=8..15, K=24..31.
__device__ __forceinline__ Raw16h raw_b_h16(const _Float16* base, int hi) {
  Raw16h r;
  r.lo = *(const v8h*)(base + 8 * hi);
  r.hi = *(const v8h*)(base + 16 + 8 * hi);
  return r;
}
__device__ __forceinline__ v16h cvt_h(Raw16h r) {
  v16h f;
#pragma unroll
  for (int i = 0; i < 8; ++i) { f[i] = r.lo[i]; f[i + 8] = r.hi[i]; }
  return f;
}
__device__ __forceinline__ Raw16f raw_a_f32(const float* base, int hi) {
  Raw16f r;
  r.a = *(const v4f*)(base + 8 * hi);
  r.b = *(const v4f*)(base + 8 * hi + 4);
  r.c = *(const v4f*)(base + 16 + 8 * hi);
  r.d = *(const v4f*)(base + 16 + 8 * hi + 4);
  return r;
}
__device__ __forceinline__ Raw16f raw_b_f32(const float* base, int hi) {
  Raw16f r;
  r.a = *(const v4f*)(base + 8 * hi);
  r.b = *(const v4f*)(base + 8 * hi + 4);
  r.c = *(const v4f*)(base + 16 + 8 * hi);
  r.d = *(const v4f*)(base + 16 + 8 * hi + 4);
  return r;
}
__device__ __forceinline__ v16h cvt_f(Raw16f r) {
  v16h f;
#pragma unroll
  for (int i = 0; i < 4; ++i) {
    f[i]      = (_Float16)r.a[i];
    f[i + 4]  = (_Float16)r.b[i];
    f[i + 8]  = (_Float16)r.c[i];
    f[i + 12] = (_Float16)r.d[i];
  }
  return f;
}

#define WMMA_F16(a, b, c) \
  __builtin_amdgcn_wmma_f32_16x16x32_f16(false, (a), false, (b), (short)0, (c), false, false)

// ---- kernel 1: fused Q/K/V/G projections (WMMA GEMM, X[4096x256] @ W.T) ----
// wave tile 32x32 (2 A-frags x 2 B-frags -> 4 accumulators), double-buffered K loop
__global__ void __launch_bounds__(256)
proj_kernel(const float* __restrict__ qx, const float* __restrict__ kx,
            const float* __restrict__ vx,
            const float* __restrict__ Wq, const float* __restrict__ Wk,
            const float* __restrict__ Wv, const float* __restrict__ Wg,
            const float* __restrict__ bg,
            _Float16* __restrict__ Qh, _Float16* __restrict__ Kh,
            _Float16* __restrict__ Vt, float* __restrict__ G) {
  const int lane = threadIdx.x & 31;
  const int wave = threadIdx.x >> 5;
  const int hi = lane >> 4;
  const int l15 = lane & 15;
  const int p = blockIdx.z;  // 0=Q 1=K 2=V 3=G
  const float* X = (p == 1) ? kx : (p == 2) ? vx : qx;
  const float* W = (p == 0) ? Wq : (p == 1) ? Wk : (p == 2) ? Wv : Wg;
  const int m0 = blockIdx.x * 128 + (wave & 3) * 32;
  const int n0 = blockIdx.y * 64 + (wave >> 2) * 32;

  const float* Xr0 = X + (size_t)(m0 + l15) * CC;
  const float* Xr1 = X + (size_t)(m0 + 16 + l15) * CC;
  const float* Wr0 = W + (size_t)(n0 + l15) * CC;
  const float* Wr1 = W + (size_t)(n0 + 16 + l15) * CC;

  v8f acc00 = {}, acc01 = {}, acc10 = {}, acc11 = {};
  Raw16f ra0 = raw_a_f32(Xr0, hi), ra1 = raw_a_f32(Xr1, hi);
  Raw16f rb0 = raw_b_f32(Wr0, hi), rb1 = raw_b_f32(Wr1, hi);
#pragma unroll
  for (int kk = 0; kk < CC; kk += 32) {
    Raw16f na0, na1, nb0, nb1;
    if (kk + 32 < CC) {  // issue next-iteration loads before consuming current
      na0 = raw_a_f32(Xr0 + kk + 32, hi);
      na1 = raw_a_f32(Xr1 + kk + 32, hi);
      nb0 = raw_b_f32(Wr0 + kk + 32, hi);
      nb1 = raw_b_f32(Wr1 + kk + 32, hi);
    }
    v16h a0 = cvt_f(ra0), a1 = cvt_f(ra1);
    v16h b0 = cvt_f(rb0), b1 = cvt_f(rb1);
    acc00 = WMMA_F16(a0, b0, acc00);
    acc01 = WMMA_F16(a0, b1, acc01);
    acc10 = WMMA_F16(a1, b0, acc10);
    acc11 = WMMA_F16(a1, b1, acc11);
    ra0 = na0; ra1 = na1; rb0 = nb0; rb1 = nb1;
  }
#pragma unroll
  for (int ai = 0; ai < 2; ++ai) {
#pragma unroll
    for (int r = 0; r < 8; ++r) {
      const int m = m0 + 16 * ai + r + 8 * hi;
      const int b = m / SS, s = m % SS;
#pragma unroll
      for (int j = 0; j < 2; ++j) {
        const int n = n0 + j * 16 + l15;
        const v8f& acc = (ai == 0) ? ((j == 0) ? acc00 : acc01)
                                   : ((j == 0) ? acc10 : acc11);
        const float val = acc[r];
        const int h = n >> 5, d = n & 31;
        if (p == 0) {
          Qh[((size_t)(b * HH + h) * SS + s) * DHD + d] = (_Float16)val;
        } else if (p == 1) {
          Kh[((size_t)(b * HH + h) * SS + s) * DHD + d] = (_Float16)val;
        } else if (p == 2) {
          Vt[((size_t)(b * HH + h) * DHD + d) * SS + s] = (_Float16)val;  // transposed
        } else {
          G[(size_t)m * CC + n] = 1.0f / (1.0f + __expf(-(val + bg[n])));
        }
      }
    }
  }
}

// ---- kernel 2: flash attention, one wave = 16 query rows, K-chunks of 32,
//      K/V/bias double-buffered across chunks -------------------------------
__global__ void __launch_bounds__(256)
attn_kernel(const _Float16* __restrict__ Qh, const _Float16* __restrict__ Kh,
            const _Float16* __restrict__ Vt, const float* __restrict__ bmask,
            const float* __restrict__ bpair, const float* __restrict__ G,
            _Float16* __restrict__ Of16) {
  __shared__ _Float16 Plds[8][16][32];  // per-wave P-transpose slab
  const int lane = threadIdx.x & 31;
  const int wave = threadIdx.x >> 5;
  const int hi = lane >> 4;
  const int l15 = lane & 15;
  const int bh = blockIdx.x;  // b*H + h
  const int b = bh >> 3;
  const int h = bh & 7;
  const int q0 = blockIdx.y * 128 + wave * 16;
  const float scale = 0.17677669529663687f;  // 1/sqrt(32)

  const _Float16* Kbase = Kh + (size_t)bh * SS * DHD;
  const _Float16* Vbase = Vt + (size_t)bh * DHD * SS;
  const float*    Mbase = bmask + (size_t)b * SS;
  const float*    Bq    = bpair + (size_t)bh * SS * SS + (size_t)q0 * SS;  // row q0

  const v16h qa = cvt_h(raw_a_h16(Qh + ((size_t)bh * SS + (q0 + l15)) * DHD, hi));
  v8f o0 = {}, o1 = {};
  float mrow[8], lrow[8];
#pragma unroll
  for (int r = 0; r < 8; ++r) { mrow[r] = -1e30f; lrow[r] = 0.0f; }

  // ---- prologue: load chunk 0 operands ----
  Raw16h rk0 = raw_b_h16(Kbase + (size_t)(0 + l15) * DHD, hi);
  Raw16h rk1 = raw_b_h16(Kbase + (size_t)(16 + l15) * DHD, hi);
  Raw16h rv0 = raw_b_h16(Vbase + (size_t)l15 * SS + 0, hi);
  Raw16h rv1 = raw_b_h16(Vbase + (size_t)(16 + l15) * SS + 0, hi);
  float bm0 = Mbase[l15], bm1 = Mbase[16 + l15];
  float bpr0[8], bpr1[8];
#pragma unroll
  for (int r = 0; r < 8; ++r) {
    const float* bp = Bq + (size_t)(r + 8 * hi) * SS;
    bpr0[r] = bp[l15];
    bpr1[r] = bp[16 + l15];
  }

  for (int kc = 0; kc < SS; kc += 32) {
    // ---- issue next chunk's loads first (latency hiding) ----
    Raw16h nk0, nk1, nv0, nv1;
    float nbm0 = 0.f, nbm1 = 0.f, nbp0[8], nbp1[8];
    const int kn = kc + 32;
    if (kn < SS) {
      nk0 = raw_b_h16(Kbase + (size_t)(kn + l15) * DHD, hi);
      nk1 = raw_b_h16(Kbase + (size_t)(kn + 16 + l15) * DHD, hi);
      nv0 = raw_b_h16(Vbase + (size_t)l15 * SS + kn, hi);
      nv1 = raw_b_h16(Vbase + (size_t)(16 + l15) * SS + kn, hi);
      nbm0 = Mbase[kn + l15];
      nbm1 = Mbase[kn + 16 + l15];
#pragma unroll
      for (int r = 0; r < 8; ++r) {
        const float* bp = Bq + (size_t)(r + 8 * hi) * SS + kn;
        nbp0[r] = bp[l15];
        nbp1[r] = bp[16 + l15];
        if (kn + 32 < SS) __builtin_prefetch(bp + 32, 0, 0);  // 2 chunks ahead
      }
    }
    // ---- QK^T ----
    v8f z = {};
    v8f s0 = WMMA_F16(qa, cvt_h(rk0), z);
    v8f s1 = WMMA_F16(qa, cvt_h(rk1), z);
    // ---- online softmax (row reductions across the 16-lane group) ----
#pragma unroll
    for (int r = 0; r < 8; ++r) {
      s0[r] = s0[r] * scale + bm0 + bpr0[r];
      s1[r] = s1[r] * scale + bm1 + bpr1[r];
      float v = fmaxf(s0[r], s1[r]);
#pragma unroll
      for (int off = 8; off > 0; off >>= 1) v = fmaxf(v, __shfl_xor(v, off, 32));
      const float mnew = fmaxf(mrow[r], v);
      const float corr = __expf(mrow[r] - mnew);
      mrow[r] = mnew;
      const float p0 = __expf(s0[r] - mnew);
      const float p1 = __expf(s1[r] - mnew);
      float rs = p0 + p1;
#pragma unroll
      for (int off = 8; off > 0; off >>= 1) rs += __shfl_xor(rs, off, 32);
      lrow[r] = lrow[r] * corr + rs;
      o0[r] *= corr;
      o1[r] *= corr;
      Plds[wave][r + 8 * hi][l15]      = (_Float16)p0;  // acc layout -> [row][key]
      Plds[wave][r + 8 * hi][16 + l15] = (_Float16)p1;
    }
    // ---- reload P as A-fragment (intra-wave LDS is in-order), P@V ----
    v16h pa = cvt_h(raw_a_h16(&Plds[wave][l15][0], hi));
    o0 = WMMA_F16(pa, cvt_h(rv0), o0);
    o1 = WMMA_F16(pa, cvt_h(rv1), o1);
    // ---- rotate double buffers ----
    rk0 = nk0; rk1 = nk1; rv0 = nv0; rv1 = nv1;
    bm0 = nbm0; bm1 = nbm1;
#pragma unroll
    for (int r = 0; r < 8; ++r) { bpr0[r] = nbp0[r]; bpr1[r] = nbp1[r]; }
  }
#pragma unroll
  for (int r = 0; r < 8; ++r) {
    const int q = q0 + r + 8 * hi;
    const size_t m = (size_t)b * SS + q;
    const float inv = 1.0f / lrow[r];
    const int c0 = h * DHD + l15;
    const int c1 = c0 + 16;
    Of16[m * CC + c0] = (_Float16)(o0[r] * inv * G[m * CC + c0]);
    Of16[m * CC + c1] = (_Float16)(o1[r] * inv * G[m * CC + c1]);
  }
}

// ---- kernel 3: output projection (O*g)[4096x256] @ Wo.T + bo -> f32 ----
__global__ void __launch_bounds__(256)
out_kernel(const _Float16* __restrict__ Of16, const float* __restrict__ Wo,
           const float* __restrict__ bo, float* __restrict__ out) {
  const int lane = threadIdx.x & 31;
  const int wave = threadIdx.x >> 5;
  const int hi = lane >> 4;
  const int l15 = lane & 15;
  const int m0 = blockIdx.x * 128 + (wave & 3) * 32;
  const int n0 = blockIdx.y * 64 + (wave >> 2) * 32;

  const _Float16* Ar0 = Of16 + (size_t)(m0 + l15) * CC;
  const _Float16* Ar1 = Of16 + (size_t)(m0 + 16 + l15) * CC;
  const float* Wr0 = Wo + (size_t)(n0 + l15) * CC;
  const float* Wr1 = Wo + (size_t)(n0 + 16 + l15) * CC;

  v8f acc00 = {}, acc01 = {}, acc10 = {}, acc11 = {};
  Raw16h ra0 = raw_a_h16(Ar0, hi), ra1 = raw_a_h16(Ar1, hi);
  Raw16f rb0 = raw_b_f32(Wr0, hi), rb1 = raw_b_f32(Wr1, hi);
#pragma unroll
  for (int kk = 0; kk < CC; kk += 32) {
    Raw16h na0, na1;
    Raw16f nb0, nb1;
    if (kk + 32 < CC) {
      na0 = raw_a_h16(Ar0 + kk + 32, hi);
      na1 = raw_a_h16(Ar1 + kk + 32, hi);
      nb0 = raw_b_f32(Wr0 + kk + 32, hi);
      nb1 = raw_b_f32(Wr1 + kk + 32, hi);
    }
    v16h a0 = cvt_h(ra0), a1 = cvt_h(ra1);
    v16h b0 = cvt_f(rb0), b1 = cvt_f(rb1);
    acc00 = WMMA_F16(a0, b0, acc00);
    acc01 = WMMA_F16(a0, b1, acc01);
    acc10 = WMMA_F16(a1, b0, acc10);
    acc11 = WMMA_F16(a1, b1, acc11);
    ra0 = na0; ra1 = na1; rb0 = nb0; rb1 = nb1;
  }
#pragma unroll
  for (int ai = 0; ai < 2; ++ai) {
#pragma unroll
    for (int r = 0; r < 8; ++r) {
      const int m = m0 + 16 * ai + r + 8 * hi;
#pragma unroll
      for (int j = 0; j < 2; ++j) {
        const int n = n0 + j * 16 + l15;
        const v8f& acc = (ai == 0) ? ((j == 0) ? acc00 : acc01)
                                   : ((j == 0) ? acc10 : acc11);
        out[(size_t)m * CC + n] = acc[r] + bo[n];
      }
    }
  }
}

extern "C" void kernel_launch(void* const* d_in, const int* in_sizes, int n_in,
                              void* d_out, int out_size, void* d_ws, size_t ws_size,
                              hipStream_t stream) {
  const float* qx    = (const float*)d_in[0];
  const float* kx    = (const float*)d_in[1];
  const float* vx    = (const float*)d_in[2];
  const float* bmask = (const float*)d_in[3];
  const float* bpair = (const float*)d_in[4];
  const float* Wq    = (const float*)d_in[5];
  const float* Wk    = (const float*)d_in[6];
  const float* Wv    = (const float*)d_in[7];
  const float* Wg    = (const float*)d_in[8];
  const float* bg    = (const float*)d_in[9];
  const float* Wo    = (const float*)d_in[10];
  const float* bo    = (const float*)d_in[11];

  const size_t nHead = (size_t)BB * HH * SS * DHD;  // 1M elems
  char* ws = (char*)d_ws;
  _Float16* Qh   = (_Float16*)ws;                          // 2 MB
  _Float16* Kh   = Qh + nHead;                             // 2 MB
  _Float16* Vt   = Kh + nHead;                             // 2 MB (transposed V)
  float*    G    = (float*)(Vt + nHead);                   // 4 MB
  _Float16* Of16 = (_Float16*)(G + (size_t)BB * SS * CC);  // 2 MB

  // 1) projections: 128x64 block tile, z = which projection
  proj_kernel<<<dim3(32, 4, 4), 256, 0, stream>>>(qx, kx, vx, Wq, Wk, Wv, Wg,
                                                  bg, Qh, Kh, Vt, G);
  // 2) flash attention: one block per (b,h, 128 query rows)
  attn_kernel<<<dim3(BB * HH, SS / 128), 256, 0, stream>>>(Qh, Kh, Vt, bmask,
                                                           bpair, G, Of16);
  // 3) output projection
  out_kernel<<<dim3(32, 4), 256, 0, stream>>>(Of16, Wo, bo, (float*)d_out);
}
